// SINGLEModel_76416058130702
// MI455X (gfx1250) — matrix operand order repaired and checked by reading, not compiled, not run on hardware
//
#include <hip/hip_runtime.h>

typedef _Float16 v16h __attribute__((ext_vector_type(16)));
typedef _Float16 h8   __attribute__((ext_vector_type(8)));
typedef float    v8f  __attribute__((ext_vector_type(8)));

static constexpr int kB  = 64;
static constexpr int kNQ = 256;
static constexpr int kNN = 512;
static constexpr int kE  = 128;   // EMB = H*D
static constexpr int kD  = 16;
static constexpr int kW  = kNN / 32;  // 16 mask words per row

// ---------------------------------------------------------------------------
// WMMA helpers (CDNA5 wave32, 16x16x32 f16 -> f32)
// ---------------------------------------------------------------------------
__device__ __forceinline__ v8f wmma_f16(v16h a, v16h b, v8f c) {
  // 8 args: (neg_a, A, neg_b, B, c_mod, C, reuse_a, reuse_b)
  return __builtin_amdgcn_wmma_f32_16x16x32_f16(false, a, false, b, (short)0, c,
                                                false, false);
}

// A operand gather from an LDS tile stored [M][32] f16 (row-major K).
// A layout used here (contiguous K, matches B packing): lane (M=L%16, half=L/16); element e: K=half*16 + e
// -> two contiguous 8-half runs at K = half*16 and K = half*16 + 8.
__device__ __forceinline__ v16h gather_a(const _Float16* rowb, int half) {
  h8 lo = *(const h8*)(rowb + half * 16);
  h8 hi = *(const h8*)(rowb + half * 16 + 8);
  return __builtin_shufflevector(lo, hi, 0, 1, 2, 3, 4, 5, 6, 7,
                                 8, 9, 10, 11, 12, 13, 14, 15);
}

// ---------------------------------------------------------------------------
// Prep kernels
// ---------------------------------------------------------------------------
// Pack a [128][128] f32 weight (row = K/input dim, col = N/output dim) into the
// per-lane WMMA B layout: tile t=(kt*8+nt), lane's 16 contiguous f16 are
// B[kt*32 + (lane/16)*16 + e][nt*16 + lane%16], e=0..15.
__global__ __launch_bounds__(32) void pack_w(const float* __restrict__ W,
                                             _Float16* __restrict__ Wp) {
  int t = blockIdx.x;            // 0..31 (kt*8+nt)
  int lane = threadIdx.x;        // 0..31
  int kt = t >> 3, nt = t & 7;
  int kh = lane >> 4, N = nt * 16 + (lane & 15);
  int k0 = kt * 32 + kh * 16;
  _Float16* dst = Wp + (size_t)(t * 32 + lane) * 16;
  for (int e = 0; e < 16; ++e)
    dst[e] = (_Float16)W[(size_t)(k0 + e) * kE + N];
}

__global__ void cvt_f16(const float* __restrict__ in, _Float16* __restrict__ o,
                        int n) {
  int i = blockIdx.x * blockDim.x + threadIdx.x;
  if (i < n) o[i] = (_Float16)in[i];
}

// Compress the 0/-inf mask into a bitmap (1 = masked) + all-inf row flags.
// One wave per (b,q) row; single pass over the 33.5 MB f32 mask.
__global__ void mask_bits_kernel(const float* __restrict__ mask,
                                 unsigned int* __restrict__ bits,
                                 int* __restrict__ flags) {
  int row = blockIdx.x * 8 + (threadIdx.x >> 5);
  int lane = threadIdx.x & 31;
  const float* p = mask + (size_t)row * kNN;
  bool allw = true;
  for (int w = 0; w < kW; ++w) {
    unsigned int word = (unsigned int)__ballot(p[w * 32 + lane] < -3.0e38f);
    allw = allw && (word == 0xffffffffu);
    if (lane == 0) bits[(size_t)row * kW + w] = word;
  }
  if (lane == 0) flags[row] = allw ? 1 : 0;
}

// ---------------------------------------------------------------------------
// Generic GEMM: Y[m, 0:128] = X[m, 0:128] @ W (+ attr[m]*wrow[n]) (+ bias[n])
// One wave per 16-row M tile; W prepacked; 32 v_wmma per wave.
// ---------------------------------------------------------------------------
__global__ __launch_bounds__(32) void gemm_xw(const _Float16* __restrict__ X,
                                              const _Float16* __restrict__ Wp,
                                              _Float16* __restrict__ Y,
                                              const float* __restrict__ attr,
                                              const float* __restrict__ wrow,
                                              const float* __restrict__ bias) {
  __shared__ _Float16 at4[16 * 128];
  const int lane = threadIdx.x;
  const int m0 = blockIdx.x * 16;
  const int M = lane & 15, half = lane >> 4;

  for (int c = 0; c < 4; ++c) {              // stage 16x128 A rows
    int ch = c * 32 + lane;
    int row = ch >> 3, seg = ch & 7;
    *(uint4*)&at4[row * 128 + seg * 8] =
        *(const uint4*)(X + (size_t)(m0 + row) * kE + seg * 8);
  }
  __syncthreads();

  v8f acc[8] = {};
  for (int kt = 0; kt < 4; ++kt) {
    v16h a = gather_a(&at4[M * 128 + kt * 32], half);
    for (int nt = 0; nt < 8; ++nt) {
      v16h bb = *(const v16h*)(Wp + (size_t)((kt * 8 + nt) * 32 + lane) * 16);
      acc[nt] = wmma_f16(a, bb, acc[nt]);
    }
  }
  for (int nt = 0; nt < 8; ++nt)
    for (int r = 0; r < 8; ++r) {
      int row = m0 + r + 8 * half;
      int col = nt * 16 + M;
      float v = acc[nt][r];
      if (attr) v += attr[row] * wrow[col];  // concat([X, attr]) @ Wq rank-1 part
      if (bias) v += bias[col];
      Y[(size_t)row * kE + col] = (_Float16)v;
    }
}

// ---------------------------------------------------------------------------
// Masked multi-head attention: one wave per (b, h, 16-query tile).
// scores kept in LDS (16x512 f32) for exact softmax; P@V via WMMA (K=512).
// Mask consumed as a bitmap staged in LDS (1 MB total vs 33.5 MB f32).
// ---------------------------------------------------------------------------
__global__ __launch_bounds__(32) void attn_mha(const _Float16* __restrict__ Qp,
                                               const _Float16* __restrict__ Kp,
                                               const _Float16* __restrict__ Vp,
                                               const unsigned int* __restrict__ bits,
                                               const int* __restrict__ allinf,
                                               _Float16* __restrict__ outc) {
  __shared__ float sc[16 * 512];
  __shared__ _Float16 at[16 * 32];
  __shared__ _Float16 bt[16 * 32];
  __shared__ unsigned int qbits[16 * kW];
  __shared__ int qfl[16];
  const int lane = threadIdx.x;
  const int qt = blockIdx.x & 15;
  const int h = (blockIdx.x >> 4) & 7;
  const int b = blockIdx.x >> 7;
  const int M = lane & 15, half = lane >> 4;
  const int q0 = qt * 16;

  // stage mask bitmap rows for this query tile (256 contiguous words)
  for (int c = 0; c < 8; ++c) {
    int idx = c * 32 + lane;
    qbits[idx] = bits[(size_t)(b * kNQ + q0) * kW + idx];
  }
  if (lane < 16) qfl[lane] = allinf[b * kNQ + q0 + lane];

  {  // stage Q tile (16 x D=16), zero-pad K to 32
    int row = lane >> 1, seg = lane & 1;
    *(uint4*)&at[row * 32 + seg * 8] =
        *(const uint4*)(Qp + (size_t)(b * kNQ + q0 + row) * kE + h * kD + seg * 8);
    uint4 z; z.x = z.y = z.z = z.w = 0u;
    *(uint4*)&at[row * 32 + 16 + seg * 8] = z;
  }
  __syncthreads();
  v16h a = gather_a(&at[M * 32], half);

  for (int mt = 0; mt < kNN / 16; ++mt) {     // scores: Q @ K^T
    __syncthreads();
    {  // B tile: bt[node][d], zero-pad K(=d) to 32
      int row = lane >> 1, seg = lane & 1;
      *(uint4*)&bt[row * 32 + seg * 8] =
          *(const uint4*)(Kp + (size_t)(b * kNN + mt * 16 + row) * kE + h * kD + seg * 8);
      uint4 z; z.x = z.y = z.z = z.w = 0u;
      *(uint4*)&bt[row * 32 + 16 + seg * 8] = z;
    }
    __syncthreads();
    v16h bb = *(const v16h*)&bt[M * 32 + half * 16];
    v8f c = {};
    c = wmma_f16(a, bb, c);
    for (int r = 0; r < 8; ++r) {
      int lr = r + 8 * half;
      int node = mt * 16 + M;
      unsigned int wb = qbits[lr * kW + (node >> 5)];
      float mv = ((wb >> (node & 31)) & 1u) ? -__builtin_inff() : 0.0f;
      if (node == 0 && qfl[lr]) mv = 0.0f;            // all-inf row fix
      sc[lr * kNN + node] = c[r] * 0.25f + mv;        // 1/sqrt(D) = 0.25
    }
  }
  __syncthreads();

  {  // softmax normalize rows in place (2 lanes per row)
    int row = lane >> 1, hc = lane & 1;
    float* p = &sc[row * kNN + hc * 256];
    float mx = -3.4e38f;
    for (int j = 0; j < 256; ++j) mx = fmaxf(mx, p[j]);
    mx = fmaxf(mx, __shfl_xor(mx, 1));
    float s = 0.f;
    for (int j = 0; j < 256; ++j) s += __expf(p[j] - mx);
    s += __shfl_xor(s, 1);
    float inv = 1.0f / s;
    for (int j = 0; j < 256; ++j) p[j] = __expf(p[j] - mx) * inv;
  }
  __syncthreads();

  v8f o = {};
  for (int kt = 0; kt < kNN / 32; ++kt) {     // out = P @ V
    __syncthreads();
    {  // stage V tile transposed: bt[d][k], k = lane
      const _Float16* vsrc = Vp + (size_t)(b * kNN + kt * 32 + lane) * kE + h * kD;
      h8 v0 = *(const h8*)(vsrc);
      h8 v1 = *(const h8*)(vsrc + 8);
      for (int d = 0; d < 8; ++d) {
        bt[d * 32 + lane] = v0[d];
        bt[(d + 8) * 32 + lane] = v1[d];
      }
    }
    __syncthreads();
    v16h pa;
    const float* prow = &sc[M * kNN + kt * 32];
    for (int e = 0; e < 16; ++e) {
      int K = half * 16 + e;
      pa[e] = (_Float16)prow[K];
    }
    v16h pb = *(const v16h*)&bt[M * 32 + half * 16];
    o = wmma_f16(pa, pb, o);
  }
  for (int r = 0; r < 8; ++r) {
    int q = q0 + r + 8 * half;
    outc[(size_t)(b * kNQ + q) * kE + h * kD + M] = (_Float16)o[r];
  }
}

// ---------------------------------------------------------------------------
// Final stage: score2 = mh @ EN^T / sqrt(128); 10*tanh + mask; softmax -> probs
// One wave per (b, 16-query tile). 128 v_wmma per wave.
// ---------------------------------------------------------------------------
__global__ __launch_bounds__(32) void final_probs(const _Float16* __restrict__ mh,
                                                  const _Float16* __restrict__ enh,
                                                  const unsigned int* __restrict__ bits,
                                                  const int* __restrict__ allinf,
                                                  float* __restrict__ out) {
  __shared__ float sc[16 * 512];
  __shared__ _Float16 at4[16 * 128];
  __shared__ _Float16 bt[16 * 32];
  __shared__ unsigned int qbits[16 * kW];
  __shared__ int qfl[16];
  const int lane = threadIdx.x;
  const int qt = blockIdx.x & 15;
  const int b = blockIdx.x >> 4;
  const int M = lane & 15, half = lane >> 4;
  const int q0 = qt * 16;

  for (int c = 0; c < 8; ++c) {               // stage mask bitmap rows
    int idx = c * 32 + lane;
    qbits[idx] = bits[(size_t)(b * kNQ + q0) * kW + idx];
  }
  if (lane < 16) qfl[lane] = allinf[b * kNQ + q0 + lane];

  for (int c = 0; c < 4; ++c) {               // stage mh A rows (16x128)
    int ch = c * 32 + lane;
    int row = ch >> 3, seg = ch & 7;
    *(uint4*)&at4[row * 128 + seg * 8] =
        *(const uint4*)(mh + (size_t)(b * kNQ + q0 + row) * kE + seg * 8);
  }
  __syncthreads();
  v16h a[4];
  for (int kt = 0; kt < 4; ++kt) a[kt] = gather_a(&at4[M * 128 + kt * 32], half);

  const float invs = 0.088388347648318447f;   // 1/sqrt(128)
  for (int nt = 0; nt < kNN / 16; ++nt) {
    v8f c = {};
    for (int kt = 0; kt < 4; ++kt) {
      __syncthreads();
      {  // B tile = EN^T: bt[node][e], e contiguous in EN row
        int row = lane >> 1, seg = lane & 1;
        const uint4* src = (const uint4*)(enh + (size_t)(b * kNN + nt * 16 + row) * kE +
                                          kt * 32 + seg * 16);
        *(uint4*)&bt[row * 32 + seg * 16] = src[0];
        *(uint4*)&bt[row * 32 + seg * 16 + 8] = src[1];
      }
      __syncthreads();
      v16h bb = *(const v16h*)&bt[M * 32 + half * 16];
      c = wmma_f16(a[kt], bb, c);
    }
    for (int r = 0; r < 8; ++r) {
      int lr = r + 8 * half;
      int node = nt * 16 + M;
      unsigned int wb = qbits[lr * kW + (node >> 5)];
      float mv = ((wb >> (node & 31)) & 1u) ? -__builtin_inff() : 0.0f;
      if (node == 0 && qfl[lr]) mv = 0.0f;
      sc[lr * kNN + node] = 10.0f * tanhf(c[r] * invs) + mv;
    }
  }
  __syncthreads();

  {  // softmax -> probs (f32 out)
    int row = lane >> 1, hc = lane & 1;
    float* p = &sc[row * kNN + hc * 256];
    float mx = -3.4e38f;
    for (int j = 0; j < 256; ++j) mx = fmaxf(mx, p[j]);
    mx = fmaxf(mx, __shfl_xor(mx, 1));
    float s = 0.f;
    for (int j = 0; j < 256; ++j) s += __expf(p[j] - mx);
    s += __shfl_xor(s, 1);
    float inv = 1.0f / s;
    float* op = out + ((size_t)b * kNQ + q0 + row) * kNN + hc * 256;
    for (int j = 0; j < 256; ++j) op[j] = __expf(p[j] - mx) * inv;
  }
}

// ---------------------------------------------------------------------------
// Host-side launcher
// ---------------------------------------------------------------------------
extern "C" void kernel_launch(void* const* d_in, const int* in_sizes, int n_in,
                              void* d_out, int out_size, void* d_ws, size_t ws_size,
                              hipStream_t stream) {
  (void)in_sizes; (void)n_in; (void)out_size; (void)ws_size;
  const float* eln  = (const float*)d_in[0];   // [B,NQ,128]
  const float* attr = (const float*)d_in[1];   // [B,NQ,1]
  const float* en   = (const float*)d_in[2];   // [B,NN,128]
  const float* mask = (const float*)d_in[3];   // [B,NQ,NN]
  const float* wq   = (const float*)d_in[4];   // [129,128]
  const float* wk   = (const float*)d_in[5];   // [128,128]
  const float* wv   = (const float*)d_in[6];   // [128,128]
  const float* wc   = (const float*)d_in[7];   // [128,128]
  const float* bc   = (const float*)d_in[8];   // [128]
  float* out = (float*)d_out;

  char* ws = (char*)d_ws;
  size_t off = 0;
  auto take = [&](size_t bytes) -> char* {
    char* p = ws + off;
    off += (bytes + 255) & ~(size_t)255;
    return p;
  };
  _Float16* wkp   = (_Float16*)take(128 * 128 * 2);
  _Float16* wvp   = (_Float16*)take(128 * 128 * 2);
  _Float16* wqp   = (_Float16*)take(128 * 128 * 2);
  _Float16* wcp   = (_Float16*)take(128 * 128 * 2);
  _Float16* enh   = (_Float16*)take((size_t)kB * kNN * kE * 2);
  _Float16* elnh  = (_Float16*)take((size_t)kB * kNQ * kE * 2);
  _Float16* kproj = (_Float16*)take((size_t)kB * kNN * kE * 2);
  _Float16* vproj = (_Float16*)take((size_t)kB * kNN * kE * 2);
  _Float16* qproj = (_Float16*)take((size_t)kB * kNQ * kE * 2);
  _Float16* outc  = (_Float16*)take((size_t)kB * kNQ * kE * 2);
  _Float16* mhbuf = (_Float16*)take((size_t)kB * kNQ * kE * 2);
  unsigned int* mbits = (unsigned int*)take((size_t)kB * kNQ * kW * 4);
  int*      flags = (int*)take((size_t)kB * kNQ * 4);

  // Prep: pack weights, convert activations, compress mask (bits + flags)
  pack_w<<<32, 32, 0, stream>>>(wk, wkp);
  pack_w<<<32, 32, 0, stream>>>(wv, wvp);
  pack_w<<<32, 32, 0, stream>>>(wq, wqp);      // rows 0..127 of [129,128]
  pack_w<<<32, 32, 0, stream>>>(wc, wcp);
  {
    int n = kB * kNN * kE;
    cvt_f16<<<(n + 255) / 256, 256, 0, stream>>>(en, enh, n);
  }
  {
    int n = kB * kNQ * kE;
    cvt_f16<<<(n + 255) / 256, 256, 0, stream>>>(eln, elnh, n);
  }
  mask_bits_kernel<<<(kB * kNQ) / 8, 256, 0, stream>>>(mask, mbits, flags);

  // Projections
  gemm_xw<<<(kB * kNN) / 16, 32, 0, stream>>>(enh, wkp, kproj, nullptr, nullptr, nullptr);
  gemm_xw<<<(kB * kNN) / 16, 32, 0, stream>>>(enh, wvp, vproj, nullptr, nullptr, nullptr);
  gemm_xw<<<(kB * kNQ) / 16, 32, 0, stream>>>(elnh, wqp, qproj, attr, wq + 128 * 128, nullptr);

  // Masked MHA -> out_concat
  attn_mha<<<kB * 8 * (kNQ / 16), 32, 0, stream>>>(qproj, kproj, vproj, mbits, flags, outc);

  // Combine: mh = out_concat @ Wcomb + bcomb
  gemm_xw<<<(kB * kNQ) / 16, 32, 0, stream>>>(outc, wcp, mhbuf, nullptr, nullptr, bc);

  // Compatibility score + tanh clip + mask + softmax -> probs
  final_probs<<<kB * (kNQ / 16), 32, 0, stream>>>(mhbuf, enh, mbits, flags, out);
}
